// FusedMoEIntegrator_51376398795154
// MI455X (gfx1250) — compile-verified, matrix operand order checked by reading, not run on hardware
//
#include <hip/hip_runtime.h>

// ---------------------------------------------------------------------------
// FusedMoEIntegrator for MI455X (gfx1250): bf16 WMMA everywhere, fused INL
// epilogues, wave32.  B=4,N=4096,D=1024 -> T=16384; E=8,K=2,H=64,ITERS=2.
// Weights kept TRANSPOSED (Bt[n][k]) so A and B WMMA fragments are contiguous
// 16-byte runs per lane (b128 loads).  LDS staging uses CDNA5 async
// global->LDS loads (ASYNCcnt) with double buffering when available.
// ---------------------------------------------------------------------------

#define TOK   16384          // B*N tokens
#define DDIM  1024
#define NE    8
#define NH    64
#define DT_C  0.1f

#if defined(__has_builtin)
#if __has_builtin(__builtin_amdgcn_global_load_async_to_lds_b128) && \
    __has_builtin(__builtin_amdgcn_global_load_async_to_lds_b64) &&  \
    __has_builtin(__builtin_amdgcn_s_wait_asynccnt)
#define USE_ASYNC_LDS 1
#endif
#endif
#ifndef USE_ASYNC_LDS
#define USE_ASYNC_LDS 0
#endif

typedef __attribute__((ext_vector_type(16))) __bf16 bf16x16;
typedef __attribute__((ext_vector_type(8)))  float  f32x8;

// ---------------- device helpers ----------------
__device__ __forceinline__ f32x8 zero8() {
  f32x8 z;
#pragma unroll
  for (int i = 0; i < 8; ++i) z[i] = 0.0f;
  return z;
}

__device__ __forceinline__ f32x8 wmma_bf16(bf16x16 a, bf16x16 b, f32x8 c) {
  // D = A(16x32 bf16) * B(32x16 bf16) + C(16x16 f32)
  return __builtin_amdgcn_wmma_f32_16x16x32_bf16(false, a, false, b,
                                                 (short)0, c, false, false);
}

#if USE_ASYNC_LDS
typedef int v4i_t __attribute__((vector_size(16)));
typedef int v2i_t __attribute__((vector_size(8)));
__device__ __forceinline__ void async_b128(const __bf16* g, __bf16* l) {
  __builtin_amdgcn_global_load_async_to_lds_b128(
      (__attribute__((address_space(1))) v4i_t*)g,
      (__attribute__((address_space(3))) v4i_t*)l, 0, 0);
}
__device__ __forceinline__ void async_b64(const __bf16* g, __bf16* l) {
  __builtin_amdgcn_global_load_async_to_lds_b64(
      (__attribute__((address_space(1))) v2i_t*)g,
      (__attribute__((address_space(3))) v2i_t*)l, 0, 0);
}
#endif

__device__ __forceinline__ float gelu_f(float x) {
  return 0.5f * x * (1.0f + erff(x * 0.70710678118654752f));
}
__device__ __forceinline__ float sigmoid_f(float x) {
  return 1.0f / (1.0f + expf(-x));
}
__device__ __forceinline__ float softplus_f(float x) {
  return (x > 20.0f) ? x : log1pf(expf(x));
}

// Row-contiguous WMMA fragment load (A tiles [16 rows x K]; transposed-B tiles
// [16 cols x K]).  lane L: row = L&15, khalf = (L>>4)*8; two contiguous
// 16-byte runs per lane -> two (ds_)load_b128.          (ISA 7.12.2)
__device__ __forceinline__ bf16x16 load_frag_row(const __bf16* base, int stride) {
  int lane = threadIdx.x & 31;
  int row  = lane & 15;
  int kh   = (lane >> 4) << 3;
  const __bf16* rp = base + (size_t)row * stride + kh;
  bf16x16 f;
#pragma unroll
  for (int i = 0; i < 8; ++i) f[i] = rp[i];
#pragma unroll
  for (int i = 0; i < 8; ++i) f[8 + i] = rp[16 + i];
  return f;
}

// ---------------- elementwise kernels ----------------
__global__ __launch_bounds__(256) void cvt_f32_bf16_kernel(
    const float* __restrict__ src, __bf16* __restrict__ dst, int n) {
  int i = blockIdx.x * 256 + threadIdx.x;
  if (i < n) dst[i] = (__bf16)src[i];
}

// src: f32 [batch, K, N] row-major  ->  dst: bf16 [batch, N, K]
__global__ __launch_bounds__(256) void cvt_transpose_kernel(
    const float* __restrict__ src, __bf16* __restrict__ dst,
    int KK, int NN, int total) {
  int i = blockIdx.x * 256 + threadIdx.x;
  if (i < total) {
    int per = KK * NN;
    int b = i / per, rem = i - b * per;
    int k = rem / NN, n = rem - k * NN;
    dst[(size_t)b * per + (size_t)n * KK + k] = (__bf16)src[i];
  }
}

__global__ __launch_bounds__(256) void init_state_kernel(
    const float* __restrict__ x, float* __restrict__ integ,
    float* __restrict__ v, int n) {
  int i = blockIdx.x * 256 + threadIdx.x;
  if (i < n) { integ[i] = x[i]; v[i] = 0.0f; }
}

__global__ __launch_bounds__(256) void build_ctx_kernel(
    const float* __restrict__ integ, const float* __restrict__ v,
    __bf16* __restrict__ ctx, int n) {
  int i = blockIdx.x * 256 + threadIdx.x;
  if (i < n) {
    int t = i >> 10, j = i & 1023;
    ctx[(size_t)t * 2048 + j]        = (__bf16)integ[i];
    ctx[(size_t)t * 2048 + 1024 + j] = (__bf16)v[i];
  }
}

__global__ __launch_bounds__(256) void blend_kernel(
    const float* __restrict__ halt, const float* __restrict__ xn,
    const float* __restrict__ vn, float* __restrict__ integ,
    float* __restrict__ v, int n) {
  int i = blockIdx.x * 256 + threadIdx.x;
  if (i < n) {
    float h = halt[i >> 10];
    integ[i] += h * (xn[i] - integ[i]);
    v[i]     += h * (vn[i] - v[i]);
  }
}

// ---------------- generic bf16 WMMA GEMM (fused bias + epilogue) ------------
// C[M,N] = epi(A[M,K] @ Bt[N,K]^T + bias).  Block tile 32(M) x 64(N), 8 waves
// as 2x4 of 16x16 WMMA tiles, K step 64 (2 WMMA per stage), double-buffered
// async LDS staging.  M%32==0, N%64==0, K%64==0 required.
enum { EPI_GELU_BF16 = 0, EPI_BF16 = 1, EPI_F32 = 2, EPI_RESID = 3 };

__global__ __launch_bounds__(256) void gemm_bf16_kernel(
    const __bf16* __restrict__ A, int lda,
    const __bf16* __restrict__ Bt,                 // [N,K] transposed weights
    const float* __restrict__ bias, void* __restrict__ out, int ldo,
    int K, int epi, const float* __restrict__ resid,
    const float* __restrict__ iw) {
  __shared__ __bf16 As[2][32][72];
  __shared__ __bf16 Bst[2][64][72];                // N-major, K-contiguous
  int tid = threadIdx.x, lane = tid & 31, wave = tid >> 5;
  int mt = wave >> 2, nt = wave & 3;
  int m0 = blockIdx.y * 32, n0 = blockIdx.x * 64;
  int rA = tid >> 3, cA = (tid & 7) * 8;           // 32 rows x 64 k (1x b128)
  int rB = tid >> 2, cB = (tid & 3) * 16;          // 64 rows x 64 k (2x b128)
  const __bf16* aG = A + (size_t)(m0 + rA) * lda + cA;
  const __bf16* bG = Bt + (size_t)(n0 + rB) * K + cB;

  auto stage = [&](int k0, int buf) {
#if USE_ASYNC_LDS
    async_b128(aG + k0, &As[buf][rA][cA]);
    async_b128(bG + k0, &Bst[buf][rB][cB]);
    async_b128(bG + k0 + 8, &Bst[buf][rB][cB + 8]);
#else
#pragma unroll
    for (int i = 0; i < 8; ++i) As[buf][rA][cA + i] = aG[k0 + i];
#pragma unroll
    for (int i = 0; i < 16; ++i) Bst[buf][rB][cB + i] = bG[k0 + i];
#endif
  };

  f32x8 acc = zero8();
  int nsteps = K >> 6;
  stage(0, 0);
  for (int s = 0; s < nsteps; ++s) {
    int cur = s & 1;
    if (s + 1 < nsteps) stage((s + 1) << 6, cur ^ 1);
#if USE_ASYNC_LDS
    if (s + 1 < nsteps) __builtin_amdgcn_s_wait_asynccnt(3);
    else                __builtin_amdgcn_s_wait_asynccnt(0);
#endif
    __syncthreads();
    bf16x16 af0 = load_frag_row(&As[cur][mt * 16][0], 72);
    bf16x16 af1 = load_frag_row(&As[cur][mt * 16][32], 72);
    bf16x16 bf0 = load_frag_row(&Bst[cur][nt * 16][0], 72);
    bf16x16 bf1 = load_frag_row(&Bst[cur][nt * 16][32], 72);
    acc = wmma_bf16(af0, bf0, acc);
    acc = wmma_bf16(af1, bf1, acc);
    __syncthreads();
  }
  int nl = lane & 15, mh = (lane >> 4) * 8;
#pragma unroll
  for (int i = 0; i < 8; ++i) {
    int gm = m0 + mt * 16 + mh + i;
    int gn = n0 + nt * 16 + nl;
    float val = acc[i] + bias[gn];
    size_t o = (size_t)gm * ldo + gn;
    if (epi == EPI_GELU_BF16)      ((__bf16*)out)[o] = (__bf16)gelu_f(val);
    else if (epi == EPI_BF16)      ((__bf16*)out)[o] = (__bf16)val;
    else if (epi == EPI_F32)       ((float*)out)[o]  = val;
    else                           ((float*)out)[o]  = resid[o] + iw[gn] * val;
  }
}

// --------- shared expert GEMM2 (64 -> 3072) fused with INL dynamics ---------
// grid = (8, T/16); each wave owns one column-triplet tile (a,b,g share j).
// W2t is [3072, 64] transposed: B fragments load straight from global (b128).
__global__ __launch_bounds__(256) void shared_g2_inl_kernel(
    const __bf16* __restrict__ Hs,                 // [T,64] bf16
    const __bf16* __restrict__ W2t,                // [3072,64] bf16
    const float* __restrict__ b2,                  // [3072]
    const float* __restrict__ integ, const float* __restrict__ v,
    const float* __restrict__ mu, const float* __restrict__ swp,
    float* __restrict__ xn_out, float* __restrict__ vn_out) {
  __shared__ __bf16 As[16][72];
  int tid = threadIdx.x, lane = tid & 31, wave = tid >> 5;
  int t0 = blockIdx.y * 16;
  {
    int r = tid >> 4, c = (tid & 15) * 4;
    const __bf16* hp = Hs + (size_t)(t0 + r) * NH + c;
#if USE_ASYNC_LDS
    async_b64(hp, &As[r][c]);
    __builtin_amdgcn_s_wait_asynccnt(0);
#else
#pragma unroll
    for (int i = 0; i < 4; ++i) As[r][c + i] = hp[i];
#endif
  }
  __syncthreads();
  int jbase = (blockIdx.x * 8 + wave) * 16;
  bf16x16 a0 = load_frag_row(&As[0][0], 72);
  bf16x16 a1 = load_frag_row(&As[0][32], 72);
  int nl = lane & 15, mh = (lane >> 4) * 8;
  float ctrl[3][8];
#pragma unroll
  for (int c = 0; c < 3; ++c) {
    int ncol = c * 1024 + jbase;
    bf16x16 b0 = load_frag_row(W2t + (size_t)ncol * NH, NH);       // k 0..31
    bf16x16 b1 = load_frag_row(W2t + (size_t)ncol * NH + 32, NH);  // k 32..63
    f32x8 acc = wmma_bf16(a0, b0, zero8());
    acc = wmma_bf16(a1, b1, acc);
    float bb = b2[ncol + nl];
#pragma unroll
    for (int i = 0; i < 8; ++i) ctrl[c][i] = acc[i] + bb;
  }
  float sw = sigmoid_f(swp[0]);
#pragma unroll
  for (int i = 0; i < 8; ++i) {
    int t = t0 + mh + i, j = jbase + nl;
    size_t o = (size_t)t * DDIM + j;
    float x = integ[o], vv = v[o], muj = mu[j];
    float vn = sigmoid_f(ctrl[0][i]) * vv - softplus_f(ctrl[1][i]) * (x - muj);
    float xn = x + DT_C * sigmoid_f(ctrl[2][i]) * vn;
    xn_out[o] = sw * xn;
    vn_out[o] = sw * vn;
  }
}

// --------- routed experts GEMM1 (2048 -> 64), dense-masked over experts -----
// grid = (T*K/32); tile 32 token-expert pairs x 64 hidden.  EW1t: [E,64,2048].
__global__ __launch_bounds__(256) void routed_g1_kernel(
    const __bf16* __restrict__ ctx,                // [T,2048]
    const int* __restrict__ topk_i,                // [T*K]
    const __bf16* __restrict__ EW1t,               // [E,64,2048] transposed
    const float* __restrict__ EB1,                 // [E,64]
    __bf16* __restrict__ HR) {                     // [T*K,64]
  __shared__ __bf16 As[32][72];
  __shared__ __bf16 Bst[64][72];
  __shared__ int sIdx[32];
  int tid = threadIdx.x, lane = tid & 31, wave = tid >> 5;
  int mt = wave >> 2, nt = wave & 3;
  int p0 = blockIdx.x * 32;
  if (tid < 32) sIdx[tid] = topk_i[p0 + tid];
  int rA = tid >> 3, cA = (tid & 7) * 8;
  int rB = tid >> 2, cB = (tid & 3) * 16;
  const __bf16* aG = ctx + (size_t)((p0 + rA) >> 1) * 2048 + cA;
  int nl = lane & 15, mh = (lane >> 4) * 8;
  for (int e = 0; e < NE; ++e) {
    const __bf16* bG = EW1t + ((size_t)e * NH + rB) * 2048 + cB;
    f32x8 acc = zero8();
    for (int k0 = 0; k0 < 2048; k0 += 64) {
#if USE_ASYNC_LDS
      async_b128(aG + k0, &As[rA][cA]);
      async_b128(bG + k0, &Bst[rB][cB]);
      async_b128(bG + k0 + 8, &Bst[rB][cB + 8]);
      __builtin_amdgcn_s_wait_asynccnt(0);
#else
#pragma unroll
      for (int i = 0; i < 8; ++i) As[rA][cA + i] = aG[k0 + i];
#pragma unroll
      for (int i = 0; i < 16; ++i) Bst[rB][cB + i] = bG[k0 + i];
#endif
      __syncthreads();
      bf16x16 af0 = load_frag_row(&As[mt * 16][0], 72);
      bf16x16 af1 = load_frag_row(&As[mt * 16][32], 72);
      bf16x16 bf0 = load_frag_row(&Bst[nt * 16][0], 72);
      bf16x16 bf1 = load_frag_row(&Bst[nt * 16][32], 72);
      acc = wmma_bf16(af0, bf0, acc);
      acc = wmma_bf16(af1, bf1, acc);
      __syncthreads();
    }
#pragma unroll
    for (int i = 0; i < 8; ++i) {
      int ml = mt * 16 + mh + i;
      if (sIdx[ml] == e) {
        int gc = nt * 16 + nl;
        float val = gelu_f(acc[i] + EB1[e * NH + gc]);
        HR[(size_t)(p0 + ml) * NH + gc] = (__bf16)val;
      }
    }
    __syncthreads();
  }
}

// --------- routed GEMM2 (64 -> 3072) + INL + top-k weighted combine ---------
// grid = (8, T*K/16); 16 pairs = 8 tokens; both K entries combined in-thread.
// EW2t: [E,3072,64] transposed; B fragments load straight from global.
__global__ __launch_bounds__(256) void routed_g2_inl_kernel(
    const __bf16* __restrict__ HR,                 // [T*K,64]
    const int* __restrict__ topk_i, const float* __restrict__ topk_w,
    const __bf16* __restrict__ EW2t,               // [E,3072,64]
    const float* __restrict__ EB2,                 // [E,3072]
    const float* __restrict__ integ, const float* __restrict__ v,
    const float* __restrict__ mu, const float* __restrict__ swp,
    float* __restrict__ xn_io, float* __restrict__ vn_io) {
  __shared__ __bf16 As[16][72];
  __shared__ int   sIdx[16];
  __shared__ float sW[16];
  int tid = threadIdx.x, lane = tid & 31, wave = tid >> 5;
  int p0 = blockIdx.y * 16;
  {
    int r = tid >> 4, c = (tid & 15) * 4;
    const __bf16* hp = HR + (size_t)(p0 + r) * NH + c;
#if USE_ASYNC_LDS
    async_b64(hp, &As[r][c]);
    __builtin_amdgcn_s_wait_asynccnt(0);
#else
#pragma unroll
    for (int i = 0; i < 4; ++i) As[r][c + i] = hp[i];
#endif
  }
  if (tid < 16) { sIdx[tid] = topk_i[p0 + tid]; sW[tid] = topk_w[p0 + tid]; }
  __syncthreads();
  int jbase = (blockIdx.x * 8 + wave) * 16;
  bf16x16 a0 = load_frag_row(&As[0][0], 72);
  bf16x16 a1 = load_frag_row(&As[0][32], 72);
  int nl = lane & 15, mh = (lane >> 4) * 8;
  float ctrl[3][8];
#pragma unroll
  for (int c = 0; c < 3; ++c)
#pragma unroll
    for (int i = 0; i < 8; ++i) ctrl[c][i] = 0.0f;

  for (int e = 0; e < NE; ++e) {
#pragma unroll
    for (int c = 0; c < 3; ++c) {
      int ncol = c * 1024 + jbase;
      const __bf16* wb = EW2t + ((size_t)e * 3072 + ncol) * NH;
      bf16x16 b0 = load_frag_row(wb, NH);
      bf16x16 b1 = load_frag_row(wb + 32, NH);
      f32x8 acc = wmma_bf16(a0, b0, zero8());
      acc = wmma_bf16(a1, b1, acc);
      float bb = EB2[(size_t)e * 3072 + ncol + nl];
#pragma unroll
      for (int i = 0; i < 8; ++i)
        if (sIdx[mh + i] == e) ctrl[c][i] = acc[i] + bb;
    }
  }
  float swc = 1.0f - sigmoid_f(swp[0]);
#pragma unroll
  for (int r = 0; r < 8; r += 2) {
    int tok = (p0 + mh + r) >> 1;
    int j = jbase + nl;
    size_t o = (size_t)tok * DDIM + j;
    float x = integ[o], vv = v[o], muj = mu[j];
    float sx = 0.0f, sv = 0.0f;
#pragma unroll
    for (int ii = 0; ii < 2; ++ii) {
      int q = r + ii;
      float vn = sigmoid_f(ctrl[0][q]) * vv - softplus_f(ctrl[1][q]) * (x - muj);
      float xn = x + DT_C * sigmoid_f(ctrl[2][q]) * vn;
      float w = sW[mh + q];
      sx += w * xn;
      sv += w * vn;
    }
    xn_io[o] += swc * sx;
    vn_io[o] += swc * sv;
  }
}

// ---------------- router head: 256 -> 8 logits, softmax, top-2 --------------
__global__ __launch_bounds__(256) void router_topk_kernel(
    const __bf16* __restrict__ RH,                 // [T,256] bf16
    const float* __restrict__ W2,                  // [256,8]
    const float* __restrict__ B2,                  // [8]
    int* __restrict__ topk_i, float* __restrict__ topk_w) {
  __shared__ float sw2[256 * 8];
  int tid = threadIdx.x;
  for (int i = tid; i < 256 * 8; i += 256) sw2[i] = W2[i];
  __syncthreads();
  int t = blockIdx.x * 256 + tid;
  float l[NE];
#pragma unroll
  for (int e = 0; e < NE; ++e) l[e] = B2[e];
  for (int i = 0; i < 256; ++i) {
    float h = (float)RH[(size_t)t * 256 + i];
#pragma unroll
    for (int e = 0; e < NE; ++e) l[e] += h * sw2[i * NE + e];
  }
  float m = l[0];
#pragma unroll
  for (int e = 1; e < NE; ++e) m = fmaxf(m, l[e]);
  float p[NE], s = 0.0f;
#pragma unroll
  for (int e = 0; e < NE; ++e) { p[e] = expf(l[e] - m); s += p[e]; }
#pragma unroll
  for (int e = 0; e < NE; ++e) p[e] /= s;
  int i0 = 0;
#pragma unroll
  for (int e = 1; e < NE; ++e) if (p[e] > p[i0]) i0 = e;
  int i1 = -1;
#pragma unroll
  for (int e = 0; e < NE; ++e)
    if (e != i0 && (i1 < 0 || p[e] > p[i1])) i1 = e;
  float ws = p[i0] + p[i1];
  topk_i[t * 2] = i0;  topk_i[t * 2 + 1] = i1;
  topk_w[t * 2] = p[i0] / ws;  topk_w[t * 2 + 1] = p[i1] / ws;
}

// ---------------- halting head: 256 -> 1, sigmoid ---------------------------
__global__ __launch_bounds__(256) void halt_kernel(
    const __bf16* __restrict__ HH,                 // [T,256] bf16 (gelu'd)
    const float* __restrict__ W2,                  // [256,1]
    const float* __restrict__ B2,                  // [1]
    float* __restrict__ halt) {
  __shared__ float sw[256];
  int tid = threadIdx.x;
  sw[tid] = W2[tid];
  __syncthreads();
  int t = blockIdx.x * 256 + tid;
  float s = B2[0];
  for (int i = 0; i < 256; ++i) s += (float)HH[(size_t)t * 256 + i] * sw[i];
  halt[t] = sigmoid_f(s);
}

// ---------------------------------------------------------------------------
extern "C" void kernel_launch(void* const* d_in, const int* in_sizes, int n_in,
                              void* d_out, int out_size, void* d_ws,
                              size_t ws_size, hipStream_t stream) {
  (void)in_sizes; (void)n_in; (void)out_size; (void)ws_size;
  const float* x    = (const float*)d_in[0];
  const float* iw   = (const float*)d_in[1];
  const float* mu   = (const float*)d_in[2];
  const float* rw1  = (const float*)d_in[3];
  const float* rb1  = (const float*)d_in[4];
  const float* rw2  = (const float*)d_in[5];
  const float* rb2  = (const float*)d_in[6];
  const float* hw1  = (const float*)d_in[7];
  const float* hb1  = (const float*)d_in[8];
  const float* hw2  = (const float*)d_in[9];
  const float* hb2  = (const float*)d_in[10];
  const float* ew1  = (const float*)d_in[11];
  const float* eb1  = (const float*)d_in[12];
  const float* ew2  = (const float*)d_in[13];
  const float* eb2  = (const float*)d_in[14];
  const float* sw1  = (const float*)d_in[15];
  const float* sb1  = (const float*)d_in[16];
  const float* sw2  = (const float*)d_in[17];
  const float* sb2  = (const float*)d_in[18];
  const float* swt  = (const float*)d_in[19];
  const float* fw1  = (const float*)d_in[20];
  const float* fb1  = (const float*)d_in[21];
  const float* fw2  = (const float*)d_in[22];
  const float* fb2  = (const float*)d_in[23];

  char* ws = (char*)d_ws;
  size_t off = 0;
  auto alloc = [&](size_t bytes) -> char* {
    char* p = ws + off;
    off += (bytes + 255) & ~(size_t)255;
    return p;
  };
  const size_t TD = (size_t)TOK * DDIM;
  __bf16* xf_bf = (__bf16*)alloc(TD * 2);
  __bf16* rw1t  = (__bf16*)alloc((size_t)256 * 1024 * 2);          // [256,1024]
  __bf16* hw1t  = (__bf16*)alloc((size_t)256 * 1024 * 2);          // [256,1024]
  __bf16* ew1t  = (__bf16*)alloc((size_t)NE * NH * 2048 * 2);      // [E,64,2048]
  __bf16* ew2t  = (__bf16*)alloc((size_t)NE * 3072 * NH * 2);      // [E,3072,64]
  __bf16* sw1t  = (__bf16*)alloc((size_t)NH * 2048 * 2);           // [64,2048]
  __bf16* sw2t  = (__bf16*)alloc((size_t)3072 * NH * 2);           // [3072,64]
  __bf16* fw1t  = (__bf16*)alloc((size_t)2048 * 1024 * 2);         // [2048,1024]
  __bf16* fw2t  = (__bf16*)alloc((size_t)1024 * 2048 * 2);         // [1024,2048]
  __bf16* hhb   = (__bf16*)alloc((size_t)TOK * 256 * 2);   // router/halt hidden
  int*    tki   = (int*)  alloc((size_t)TOK * 2 * 4);
  float*  tkw   = (float*)alloc((size_t)TOK * 2 * 4);
  float*  integ = (float*)alloc(TD * 4);
  float*  vbuf  = (float*)alloc(TD * 4);
  __bf16* ctx   = (__bf16*)alloc((size_t)TOK * 2048 * 2);
  __bf16* hsh   = (__bf16*)alloc((size_t)TOK * NH * 2);
  __bf16* hr    = (__bf16*)alloc((size_t)TOK * 2 * NH * 2);
  float*  xn    = (float*)alloc(TD * 4);                   // also refine hidden
  float*  vn    = (float*)alloc(TD * 4);
  float*  halt  = (float*)alloc((size_t)TOK * 4);
  __bf16* r1    = (__bf16*)xn;                             // alias: [T,2048] bf16

  auto cvt_t = [&](const float* s, __bf16* d, int KK, int NN, int batch) {
    int total = batch * KK * NN;
    cvt_transpose_kernel<<<(total + 255) / 256, 256, 0, stream>>>(s, d, KK, NN,
                                                                  total);
  };
  // weight conversions + transposes (cheap; rerun every call for determinism)
  cvt_t(rw1, rw1t, 1024, 256, 1);
  cvt_t(hw1, hw1t, 1024, 256, 1);
  cvt_t(ew1, ew1t, 2048, NH, NE);
  cvt_t(ew2, ew2t, NH, 3072, NE);
  cvt_t(sw1, sw1t, 2048, NH, 1);
  cvt_t(sw2, sw2t, NH, 3072, 1);
  cvt_t(fw1, fw1t, 1024, 2048, 1);
  cvt_t(fw2, fw2t, 2048, 1024, 1);
  cvt_f32_bf16_kernel<<<(unsigned)(TD / 256), 256, 0, stream>>>(x, xf_bf,
                                                                (int)TD);

  const unsigned EW_GRID = (unsigned)(TD / 256);
  init_state_kernel<<<EW_GRID, 256, 0, stream>>>(x, integ, vbuf, (int)TD);

  // Router: GEMM1 (gelu->bf16) then logits/softmax/top-2
  gemm_bf16_kernel<<<dim3(256 / 64, TOK / 32), 256, 0, stream>>>(
      xf_bf, 1024, rw1t, rb1, hhb, 256, 1024, EPI_GELU_BF16, nullptr, nullptr);
  router_topk_kernel<<<TOK / 256, 256, 0, stream>>>(hhb, rw2, rb2, tki, tkw);

  for (int it = 0; it < 2; ++it) {
    build_ctx_kernel<<<EW_GRID, 256, 0, stream>>>(integ, vbuf, ctx, (int)TD);
    // shared expert GEMM1: ctx[T,2048] @ sw1[2048,64] -> gelu -> hsh
    gemm_bf16_kernel<<<dim3(NH / 64, TOK / 32), 256, 0, stream>>>(
        ctx, 2048, sw1t, sb1, hsh, NH, 2048, EPI_GELU_BF16, nullptr, nullptr);
    // shared GEMM2 + INL -> xn/vn = sw * (x_sh, v_sh)
    shared_g2_inl_kernel<<<dim3(8, TOK / 16), 256, 0, stream>>>(
        hsh, sw2t, sb2, integ, vbuf, mu, swt, xn, vn);
    // routed experts GEMM1 (dense-masked) -> hr
    routed_g1_kernel<<<(TOK * 2) / 32, 256, 0, stream>>>(ctx, tki, ew1t, eb1,
                                                         hr);
    // routed GEMM2 + INL + top-k combine -> xn/vn += (1-sw)*(x_r, v_r)
    routed_g2_inl_kernel<<<dim3(8, (TOK * 2) / 16), 256, 0, stream>>>(
        hr, tki, tkw, ew2t, eb2, integ, vbuf, mu, swt, xn, vn);
    // halting head: GEMM1 over first 1024 cols of ctx (lda=2048)
    gemm_bf16_kernel<<<dim3(256 / 64, TOK / 32), 256, 0, stream>>>(
        ctx, 2048, hw1t, hb1, hhb, 256, 1024, EPI_GELU_BF16, nullptr, nullptr);
    halt_kernel<<<TOK / 256, 256, 0, stream>>>(hhb, hw2, hb2, halt);
    blend_kernel<<<EW_GRID, 256, 0, stream>>>(halt, xn, vn, integ, vbuf,
                                              (int)TD);
  }

  // Refine: rebuild ctx (bf16 integrated lives in its first 1024 cols)
  build_ctx_kernel<<<EW_GRID, 256, 0, stream>>>(integ, vbuf, ctx, (int)TD);
  gemm_bf16_kernel<<<dim3(2048 / 64, TOK / 32), 256, 0, stream>>>(
      ctx, 2048, fw1t, fb1, r1, 2048, 1024, EPI_GELU_BF16, nullptr, nullptr);
  // final GEMM fused with residual: out = x + iw * (r1 @ fw2 + fb2)
  gemm_bf16_kernel<<<dim3(1024 / 64, TOK / 32), 256, 0, stream>>>(
      r1, 2048, fw2t, fb2, d_out, 1024, 2048, EPI_RESID, x, iw);
}
